// ScannedRNN_42528766165022
// MI455X (gfx1250) — matrix operand order repaired and checked by reading, not compiled, not run on hardware
//
#include <hip/hip_runtime.h>
#include <stdint.h>

typedef __attribute__((ext_vector_type(16))) _Float16 v16h;
typedef __attribute__((ext_vector_type(8)))  float    v8f;

#define T_DIM 4096
#define B_DIM 2048
#define F_DIM 10
#define H_DIM 10

union HV { uint32_t u[8]; v16h v; };

__device__ __forceinline__ uint32_t packh2(float a, float b) {
  _Float16 lo = (_Float16)a, hi = (_Float16)b;
  uint16_t ul = __builtin_bit_cast(uint16_t, lo);
  uint16_t uh = __builtin_bit_cast(uint16_t, hi);
  return (uint32_t)ul | ((uint32_t)uh << 16);
}

__device__ __forceinline__ float fsig(float u) {
  float e = __expf(-u);
  return __builtin_amdgcn_rcpf(1.0f + e);
}
__device__ __forceinline__ float ftanh(float u) {
  float e = __expf(-2.0f * u);
  return (1.0f - e) * __builtin_amdgcn_rcpf(1.0f + e);
}

// K-index remap for the B operand (avoids all cross-lane shuffles):
//   K 0..7   -> x[0..7]        (lower-lane region, vgprs 0..3)
//   K 8..15  -> h[0..7]        (lower-lane region, vgprs 4..7)
//   K 16,17  -> x[8,9]         (upper-lane region, vgpr 0)
//   K 18,19  -> h[8,9]         (upper-lane region, vgpr 1)
//   K 20..31 -> 0
__device__ __forceinline__ void kmap(int k, int& src /*0=Wi,1=Wh,2=none*/, int& row) {
  if (k < 8)           { src = 0; row = k;      return; }
  if (k < 16)          { src = 1; row = k - 8;  return; }
  if (k == 16 || k == 17) { src = 0; row = k - 8;  return; }  // x8, x9
  if (k == 18 || k == 19) { src = 1; row = k - 10; return; }  // h8, h9
  src = 2; row = 0;
}

__global__ __launch_bounds__(32) void gru_scan_wmma(
    const float* __restrict__ x, const unsigned char* __restrict__ resets,
    const float* __restrict__ Wi, const float* __restrict__ Wh,
    const float* __restrict__ bi, const float* __restrict__ bhn,
    float* __restrict__ out)
{
  const int lane = threadIdx.x;          // wave32
  const int j    = lane & 15;            // A/D row index (M) and batch col (N)
  const int hi   = lane >> 4;            // half-wave select
  const int b0   = blockIdx.x * 16;
  const int br   = b0 + j;               // batch row owned by this lane

  // ---- loop-invariant A tiles (weights, transposed, K remapped) ----
  // K order used here (B matches): vgpr v -> K = 16*hi+2v (+1)
  HV Ar, Az, Ain, Ahn;
  #pragma unroll
  for (int v = 0; v < 8; ++v) {
    int k0 = 16*hi + 2*v;
    float e[8] = {0,0,0,0,0,0,0,0};   // {r0,r1, z0,z1, in0,in1, hn0,hn1}
    if (j < H_DIM) {
      #pragma unroll
      for (int p = 0; p < 2; ++p) {
        int src, row; kmap(k0 + p, src, row);
        if (src == 0) {               // Wi row
          e[0+p] = Wi[row * (3*H_DIM) + j];
          e[2+p] = Wi[row * (3*H_DIM) + H_DIM + j];
          e[4+p] = Wi[row * (3*H_DIM) + 2*H_DIM + j];
        } else if (src == 1) {        // Wh row
          e[0+p] = Wh[row * (3*H_DIM) + j];
          e[2+p] = Wh[row * (3*H_DIM) + H_DIM + j];
          e[6+p] = Wh[row * (3*H_DIM) + 2*H_DIM + j];
        }
      }
    }
    Ar.u[v]  = packh2(e[0], e[1]);
    Az.u[v]  = packh2(e[2], e[3]);
    Ain.u[v] = packh2(e[4], e[5]);
    Ahn.u[v] = packh2(e[6], e[7]);
  }

  // ---- biases as WMMA C operands (C layout: row = v + 8*hi, const across lanes) ----
  v8f Cr, Cz, Cin, Chn;
  #pragma unroll
  for (int v = 0; v < 8; ++v) {
    int row = v + 8*hi;
    bool ok = row < H_DIM;
    Cr[v]  = ok ? bi[row]           : 0.0f;
    Cz[v]  = ok ? bi[H_DIM + row]   : 0.0f;
    Cin[v] = ok ? bi[2*H_DIM + row] : 0.0f;
    Chn[v] = ok ? bhn[row]          : 0.0f;
  }

  // ---- scan state: h in D layout (row = v+8*hi, col = batch lane) ----
  float Hprev[8];
  #pragma unroll
  for (int v = 0; v < 8; ++v) Hprev[v] = 0.0f;

  const float2* xp = (const float2*)x + (size_t)br * (F_DIM/2);
  const size_t  xstride = (size_t)B_DIM * (F_DIM/2);     // float2 per timestep
  const unsigned char* rp = resets + br;
  float* op = out + (size_t)br * H_DIM;

  // software prefetch of t=0
  float2 nx0 = xp[0], nx1 = xp[1], nx2 = xp[2], nx3 = xp[3], nx4 = xp[4];
  unsigned char nr = rp[0];

  const bool lowhalf = (hi == 0);

  for (int t = 0; t < T_DIM; ++t) {
    float2 x01 = nx0, x23 = nx1, x45 = nx2, x67 = nx3, x89 = nx4;
    const bool rst = (nr != 0);
    if (t + 1 < T_DIM) {                     // prefetch next step early
      const float2* xq = xp + (size_t)(t+1) * xstride;
      nx0 = xq[0]; nx1 = xq[1]; nx2 = xq[2]; nx3 = xq[3]; nx4 = xq[4];
      nr  = rp[(size_t)(t+1) * B_DIM];
    }

    // reset-masked carry
    float hm[8];
    #pragma unroll
    for (int v = 0; v < 8; ++v) hm[v] = rst ? 0.0f : Hprev[v];

    uint32_t p01 = packh2(hm[0], hm[1]);     // lower: (h0,h1); upper: (h8,h9)

    // B tile, remapped K (no shuffles needed)
    HV Bv;
    Bv.u[0] = lowhalf ? packh2(x01.x, x01.y) : packh2(x89.x, x89.y); // K0,1 | K16,17
    Bv.u[1] = lowhalf ? packh2(x23.x, x23.y) : p01;                  // K2,3 | K18,19
    Bv.u[2] = lowhalf ? packh2(x45.x, x45.y) : 0u;                   // K4,5 | 0
    Bv.u[3] = lowhalf ? packh2(x67.x, x67.y) : 0u;                   // K6,7 | 0
    Bv.u[4] = lowhalf ? p01 : 0u;                                    // K8,9 | 0
    Bv.u[5] = packh2(hm[2], hm[3]);  // K10,11 | rows 10,11 == 0 naturally
    Bv.u[6] = packh2(hm[4], hm[5]);  // K12,13 | rows 12,13 == 0 naturally
    Bv.u[7] = packh2(hm[6], hm[7]);  // K14,15 | rows 14,15 == 0 naturally

    // Four independent WMMAs, biases pre-loaded via C operand
    v8f Dr  = __builtin_amdgcn_wmma_f32_16x16x32_f16(false, Ar.v,  false, Bv.v, (short)0, Cr,  false, false);
    v8f Dz  = __builtin_amdgcn_wmma_f32_16x16x32_f16(false, Az.v,  false, Bv.v, (short)0, Cz,  false, false);
    v8f Din = __builtin_amdgcn_wmma_f32_16x16x32_f16(false, Ain.v, false, Bv.v, (short)0, Cin, false, false);
    v8f Dhn = __builtin_amdgcn_wmma_f32_16x16x32_f16(false, Ahn.v, false, Bv.v, (short)0, Chn, false, false);

    // gate math, pure per-lane in shared D layout
    float Hn[8];
    #pragma unroll
    for (int v = 0; v < 8; ++v) {
      float r = fsig(Dr[v]);
      float z = fsig(Dz[v]);
      float n = ftanh(Din[v] + r * Dhn[v]);
      Hn[v] = n + z * (hm[v] - n);           // (1-z)*n + z*h
      Hprev[v] = Hn[v];
    }

    // store ys[t, br, 0..9]; rows 0-7 in lower half, rows 8,9 in upper
    float* o = op + (size_t)t * (B_DIM * H_DIM);
    if (lowhalf) {
      *(float2*)(o + 0) = make_float2(Hn[0], Hn[1]);
      *(float2*)(o + 2) = make_float2(Hn[2], Hn[3]);
      *(float2*)(o + 4) = make_float2(Hn[4], Hn[5]);
      *(float2*)(o + 6) = make_float2(Hn[6], Hn[7]);
    } else {
      *(float2*)(o + 8) = make_float2(Hn[0], Hn[1]);   // rows 8,9
    }
  }
}

extern "C" void kernel_launch(void* const* d_in, const int* in_sizes, int n_in,
                              void* d_out, int out_size, void* d_ws, size_t ws_size,
                              hipStream_t stream) {
  const float*         x      = (const float*)d_in[0];
  const unsigned char* resets = (const unsigned char*)d_in[1];
  const float*         Wi     = (const float*)d_in[2];
  const float*         Wh     = (const float*)d_in[3];
  const float*         bi     = (const float*)d_in[4];
  const float*         bhn    = (const float*)d_in[5];
  float*               out    = (float*)d_out;
  (void)in_sizes; (void)n_in; (void)out_size; (void)d_ws; (void)ws_size;

  dim3 grid(B_DIM / 16);   // 128 independent 16-batch scan chains
  dim3 block(32);          // one wave32 per chain
  hipLaunchKernelGGL(gru_scan_wmma, grid, block, 0, stream,
                     x, resets, Wi, Wh, bi, bhn, out);
}
